// SequencePredictor_56255481643257
// MI455X (gfx1250) — compile-verified, matrix operand order checked by reading, not run on hardware
//
#include <hip/hip_runtime.h>
#include <hip/hip_bf16.h>
#include <math.h>

typedef __attribute__((ext_vector_type(16))) _Float16 v16h;
typedef __attribute__((ext_vector_type(8)))  float    v8f;
typedef __attribute__((ext_vector_type(4)))  int      v4i;

// Problem constants
#define Bb   32
#define TX   128
#define TY   128
#define Vv   32000
#define Dd   256
#define Hh   512
#define G4H  2048           // 4*H
#define NT_G 128            // N tiles of gate matrix (2048/16)
#define KT_X 8              // K tiles for x GEMM (256/32)
#define KT_H 16             // K tiles for h GEMM (512/32)
#define NT_V 2000           // N tiles of vocab (32000/16)

// ---- CDNA5 async-copy path (guarded; falls back to sync LDS staging) -------
#if defined(__has_builtin)
#if __has_builtin(__builtin_amdgcn_s_wait_asynccnt)
#define WAIT_ASYNC0() __builtin_amdgcn_s_wait_asynccnt(0)
#else
#define WAIT_ASYNC0() asm volatile("s_wait_asynccnt 0x0" ::: "memory")
#endif
#if __has_builtin(__builtin_amdgcn_global_load_async_to_lds_b128)
#define HAVE_ASYNC_LDS 1
#endif
#else
#define WAIT_ASYNC0() asm volatile("s_wait_asynccnt 0x0" ::: "memory")
#endif

__device__ __forceinline__ void async_copy16B(const _Float16* gsrc, _Float16* ldst) {
#ifdef HAVE_ASYNC_LDS
    __builtin_amdgcn_global_load_async_to_lds_b128(
        (__attribute__((address_space(1))) v4i*)gsrc,
        (__attribute__((address_space(3))) v4i*)ldst, 0, 0);
#else
    *(uint4*)ldst = *(const uint4*)gsrc;
#endif
}

// WMMA 16x16x32 f16 operand swizzle: per-lane half index e (0..15), lane l (0..31)
__device__ __forceinline__ int klane(int l, int e) {
    return (e & 7) + 8 * (2 * (e >> 3) + (l >> 4));
}
__device__ __forceinline__ float sigmoidf_(float x) {
    return 1.0f / (1.0f + __expf(-x));
}

// ---------------------------------------------------------------------------
// Pack embedded tokens into WMMA-A layout, f32 -> f16.
// A rows ordered row = t*32 + b ; tile addressing ((mt*Ktiles + kt)*512 + lane*16 + e)
// ---------------------------------------------------------------------------
__global__ void pack_x_kernel(const int* __restrict__ ids, const float* __restrict__ E,
                              _Float16* __restrict__ dst, int T, int total) {
    int idx = blockIdx.x * blockDim.x + threadIdx.x;
    if (idx >= total) return;
    int e    = idx & 15;
    int lane = (idx >> 4) & 31;
    int tile = idx >> 9;
    int kt   = tile % KT_X;
    int mt   = tile / KT_X;
    int row  = 16 * mt + (lane & 15);      // = t*32 + b
    int t    = row >> 5;
    int b    = row & 31;
    int d    = 32 * kt + klane(lane, e);
    int tok  = ids[b * T + t];
    dst[idx] = (_Float16)E[(size_t)tok * Dd + d];
}

// ---------------------------------------------------------------------------
// Pack a weight matrix [K x N] (row-major, stride srcStride) into WMMA-B layout.
// Tile addressing ((kt*Ntiles + nt)*512 + lane*16 + e)
// ---------------------------------------------------------------------------
__global__ void pack_b_kernel(const float* __restrict__ src, _Float16* __restrict__ dst,
                              int Ntiles, int srcStride, int total) {
    int idx = blockIdx.x * blockDim.x + threadIdx.x;
    if (idx >= total) return;
    int e    = idx & 15;
    int lane = (idx >> 4) & 31;
    int tile = idx >> 9;
    int nt   = tile % Ntiles;
    int kt   = tile / Ntiles;
    int K    = 32 * kt + klane(lane, e);
    int N    = 16 * nt + (lane & 15);
    dst[idx] = (_Float16)src[(size_t)K * srcStride + N];
}

// ---------------------------------------------------------------------------
// pre[t] = x_t @ Wx + b ; output in D-tile layout ((mt*NT_G + nt)*256 + lane*8).
// One wave computes 4 adjacent N tiles for one M tile; K loop double-buffered.
// ---------------------------------------------------------------------------
__global__ __launch_bounds__(256) void gemm_pre_kernel(
    const _Float16* __restrict__ Apack, const _Float16* __restrict__ Bpack,
    const float* __restrict__ bias, float* __restrict__ preOut, int Mtiles) {
    int gw   = blockIdx.x * 8 + (threadIdx.x >> 5);
    int lane = threadIdx.x & 31;
    int mt   = gw >> 5;        // /32
    int nb   = gw & 31;        // 4 n-tiles each
    if (mt >= Mtiles) return;

    const _Float16* aBase = Apack + (size_t)mt * KT_X * 512 + lane * 16;
    const _Float16* bBase = Bpack + (size_t)(nb * 4) * 512 + lane * 16;

    v8f acc[4];
#pragma unroll
    for (int q = 0; q < 4; ++q) {
        float bv = bias[(nb * 4 + q) * 16 + (lane & 15)];
#pragma unroll
        for (int r = 0; r < 8; ++r) acc[q][r] = bv;
    }

    v16h a = *(const v16h*)aBase;
    v16h bcur[4];
#pragma unroll
    for (int q = 0; q < 4; ++q) bcur[q] = *(const v16h*)(bBase + (size_t)q * 512);

#pragma unroll
    for (int kt = 0; kt < KT_X; ++kt) {
        v16h anx;
        v16h bnx[4];
        if (kt + 1 < KT_X) {
            anx = *(const v16h*)(aBase + (size_t)(kt + 1) * 512);
#pragma unroll
            for (int q = 0; q < 4; ++q)
                bnx[q] = *(const v16h*)(bBase + ((size_t)(kt + 1) * NT_G + q) * 512);
        }
#pragma unroll
        for (int q = 0; q < 4; ++q)
            acc[q] = __builtin_amdgcn_wmma_f32_16x16x32_f16(false, a, false, bcur[q],
                                                            (short)0, acc[q], false, false);
        if (kt + 1 < KT_X) {
            a = anx;
#pragma unroll
            for (int q = 0; q < 4; ++q) bcur[q] = bnx[q];
        }
    }
#pragma unroll
    for (int q = 0; q < 4; ++q)
        *(v8f*)(preOut + ((size_t)mt * NT_G + nb * 4 + q) * 256 + lane * 8) = acc[q];
}

// ---------------------------------------------------------------------------
// Persistent recurrence: one workgroup (32 waves) runs encoder then decoder.
// h kept in LDS pre-swizzled as WMMA-A tiles; c kept in VGPRs (wave owns fixed units).
// Decoder h also scattered to global dech in projection-A layout (row = t*32+b).
// ---------------------------------------------------------------------------
__global__ __launch_bounds__(1024) void lstm_seq_kernel(
    const float* __restrict__ pre_enc, const float* __restrict__ pre_dec,
    const _Float16* __restrict__ whe, const _Float16* __restrict__ whd,
    _Float16* __restrict__ dech, int Tenc, int Tdec) {
    __shared__ _Float16 hlds[2 * KT_H * 512];   // 32KB, A-tile layout (mt,kt)

    int tid  = threadIdx.x;
    int lane = tid & 31;
    int w    = tid >> 5;

    for (int i = tid; i < 2 * KT_H * 512; i += 1024) hlds[i] = (_Float16)0.f;
    __syncthreads();

    v8f cst[2] = {};            // c state: one 16x16 tile-column per unit
    _Float16 hn[2][8];

    for (int phase = 0; phase < 2; ++phase) {
        const float*    pre = phase ? pre_dec : pre_enc;
        const _Float16* wh  = phase ? whd : whe;
        int T = phase ? Tdec : Tenc;

        for (int t = 0; t < T; ++t) {
            const float* preT = pre + (size_t)t * 2 * NT_G * 256;
#pragma unroll
            for (int ui = 0; ui < 2; ++ui) {
                int u  = 2 * w + ui;
                int mt = u >> 5;
                int nt = u & 31;
                const _Float16* whB = wh + (size_t)nt * 512 + lane * 16;
                const _Float16* aB  = hlds + (size_t)mt * KT_H * 512 + lane * 16;

                v8f ai = *(const v8f*)(preT + ((size_t)(mt * NT_G + nt))      * 256 + lane * 8);
                v8f aj = *(const v8f*)(preT + ((size_t)(mt * NT_G + nt + 32)) * 256 + lane * 8);
                v8f af = *(const v8f*)(preT + ((size_t)(mt * NT_G + nt + 64)) * 256 + lane * 8);
                v8f ao = *(const v8f*)(preT + ((size_t)(mt * NT_G + nt + 96)) * 256 + lane * 8);

                v16h a = *(const v16h*)aB;
                v16h bc[4];
#pragma unroll
                for (int g = 0; g < 4; ++g)
                    bc[g] = *(const v16h*)(whB + (size_t)(g * 32) * 512);

#pragma unroll
                for (int kt = 0; kt < KT_H; ++kt) {
                    v16h an;
                    v16h bn[4];
                    if (kt + 1 < KT_H) {
                        an = *(const v16h*)(aB + (size_t)(kt + 1) * 512);
#pragma unroll
                        for (int g = 0; g < 4; ++g)
                            bn[g] = *(const v16h*)(whB + ((size_t)(kt + 1) * NT_G + g * 32) * 512);
                    }
                    ai = __builtin_amdgcn_wmma_f32_16x16x32_f16(false, a, false, bc[0], (short)0, ai, false, false);
                    aj = __builtin_amdgcn_wmma_f32_16x16x32_f16(false, a, false, bc[1], (short)0, aj, false, false);
                    af = __builtin_amdgcn_wmma_f32_16x16x32_f16(false, a, false, bc[2], (short)0, af, false, false);
                    ao = __builtin_amdgcn_wmma_f32_16x16x32_f16(false, a, false, bc[3], (short)0, ao, false, false);
                    if (kt + 1 < KT_H) {
                        a = an;
#pragma unroll
                        for (int g = 0; g < 4; ++g) bc[g] = bn[g];
                    }
                }
#pragma unroll
                for (int r = 0; r < 8; ++r) {
                    float cn = sigmoidf_(af[r] + 1.0f) * cst[ui][r]
                             + sigmoidf_(ai[r]) * tanhf(aj[r]);
                    cst[ui][r] = cn;
                    hn[ui][r]  = (_Float16)(sigmoidf_(ao[r]) * tanhf(cn));
                }
            }
            __syncthreads();            // all waves done reading old h
#pragma unroll
            for (int ui = 0; ui < 2; ++ui) {
                int u   = 2 * w + ui;
                int mt  = u >> 5;
                int nt  = u & 31;
                int col = 16 * nt + (lane & 15);   // h-dim column, fixed per lane
                int kt  = col >> 5;
                int Kp  = col & 31;
                int bq  = Kp >> 3;
                int ea  = (Kp & 7) + 8 * (bq >> 1);
#pragma unroll
                for (int r = 0; r < 8; ++r) {
                    int Mp = r + 8 * (lane >> 4);
                    int la = Mp + 16 * (bq & 1);
                    hlds[(size_t)(mt * KT_H + kt) * 512 + la * 16 + ea] = hn[ui][r];
                    if (phase == 1) {
                        int mtp = t * 2 + mt;      // projection A row-tile (row = t*32+b)
                        dech[(size_t)(mtp * KT_H + kt) * 512 + la * 16 + ea] = hn[ui][r];
                    }
                }
            }
            __syncthreads();            // new h visible before next step
        }
    }
}

// ---------------------------------------------------------------------------
// preds = dech @ W_out + b_out -> d_out [B, TY, V] f32.
// Block = 8 M-tiles x 4 shared N tiles. B tiles async-copied global->LDS into a
// double buffer shared by all 8 waves (8x L2 traffic cut); waves pipeline
// WMMAs on stage k against the async fill of stage k+1 (ASYNCcnt-decoupled).
// ---------------------------------------------------------------------------
__global__ __launch_bounds__(256) void proj_kernel(
    const _Float16* __restrict__ Apack, const _Float16* __restrict__ Bpack,
    const float* __restrict__ bias, float* __restrict__ out) {
    __shared__ alignas(32) _Float16 bstage[2 * 4 * 512];   // 8KB double-buffered B stage

    int tid  = threadIdx.x;
    int lane = tid & 31;
    int w    = tid >> 5;
    int mtb  = blockIdx.x / 500;     // 32 M-blocks of 8 tiles
    int nb   = blockIdx.x % 500;     // 4 N-tiles per block
    int mt   = mtb * 8 + w;

    const _Float16* aBase = Apack + (size_t)mt * KT_H * 512 + lane * 16;
    const _Float16* bBase = Bpack + (size_t)(nb * 4) * 512;  // 4 tiles contiguous = 4KB/stage

    v8f acc[4];
#pragma unroll
    for (int q = 0; q < 4; ++q) {
        float bv = bias[(nb * 4 + q) * 16 + (lane & 15)];
#pragma unroll
        for (int r = 0; r < 8; ++r) acc[q][r] = bv;
    }

    // Stage 0: 4KB cooperative copy, 16B per thread
    async_copy16B(bBase + (size_t)tid * 8, &bstage[tid * 8]);
    v16h a = *(const v16h*)aBase;
    WAIT_ASYNC0();
    __syncthreads();

#pragma unroll
    for (int kt = 0; kt < KT_H; ++kt) {
        int cur = kt & 1;
        if (kt + 1 < KT_H)          // launch next stage while computing this one
            async_copy16B(bBase + (size_t)(kt + 1) * NT_V * 512 + tid * 8,
                          &bstage[(cur ^ 1) * 2048 + tid * 8]);
        v16h anx;
        if (kt + 1 < KT_H)
            anx = *(const v16h*)(aBase + (size_t)(kt + 1) * 512);

        v16h bt[4];
#pragma unroll
        for (int q = 0; q < 4; ++q)
            bt[q] = *(const v16h*)(&bstage[cur * 2048 + q * 512 + lane * 16]);
#pragma unroll
        for (int q = 0; q < 4; ++q)
            acc[q] = __builtin_amdgcn_wmma_f32_16x16x32_f16(false, a, false, bt[q],
                                                            (short)0, acc[q], false, false);
        if (kt + 1 < KT_H) {
            a = anx;
            WAIT_ASYNC0();          // own async copy for stage kt+1 landed
            __syncthreads();        // everyone's copy landed; stage buffers safe to swap
        }
    }
#pragma unroll
    for (int q = 0; q < 4; ++q) {
        int v = (nb * 4 + q) * 16 + (lane & 15);
#pragma unroll
        for (int r = 0; r < 8; ++r) {
            int row = 16 * mt + r + 8 * (lane >> 4);   // row = t*32 + b
            int t   = row >> 5;
            int b   = row & 31;
            out[((size_t)b * TY + t) * Vv + v] = acc[q][r];
        }
    }
}

// ---------------------------------------------------------------------------
extern "C" void kernel_launch(void* const* d_in, const int* in_sizes, int n_in,
                              void* d_out, int out_size, void* d_ws, size_t ws_size,
                              hipStream_t stream) {
    (void)in_sizes; (void)n_in; (void)out_size; (void)ws_size;
    const int*   enc_ids = (const int*)d_in[0];
    const int*   dec_ids = (const int*)d_in[1];
    const float* E       = (const float*)d_in[2];
    const float* W_enc   = (const float*)d_in[3];
    const float* b_enc   = (const float*)d_in[4];
    const float* W_dec   = (const float*)d_in[5];
    const float* b_dec   = (const float*)d_in[6];
    const float* W_out   = (const float*)d_in[7];
    const float* b_out   = (const float*)d_in[8];
    float* out = (float*)d_out;

    // Workspace carve-up (256B aligned)
    char* ws = (char*)d_ws;
    size_t off = 0;
    auto alloc = [&](size_t bytes) { void* p = ws + off; off = (off + bytes + 255) & ~(size_t)255; return p; };
    const size_t XE_H  = (size_t)256 * KT_X * 512;     // activation A halves (per phase)
    const size_t WX_H  = (size_t)KT_X * NT_G * 512;
    const size_t WH_H  = (size_t)KT_H * NT_G * 512;
    const size_t WO_H  = (size_t)KT_H * NT_V * 512;
    const size_t DH_H  = (size_t)256 * KT_H * 512;
    const size_t PRE_F = (size_t)TX * 2 * NT_G * 256;  // f32 per phase (8.4M)

    _Float16* xe    = (_Float16*)alloc(XE_H * 2);
    _Float16* xd    = (_Float16*)alloc(XE_H * 2);
    _Float16* wxe   = (_Float16*)alloc(WX_H * 2);
    _Float16* whe   = (_Float16*)alloc(WH_H * 2);
    _Float16* wxd   = (_Float16*)alloc(WX_H * 2);
    _Float16* whd   = (_Float16*)alloc(WH_H * 2);
    _Float16* wout  = (_Float16*)alloc(WO_H * 2);
    _Float16* dech  = (_Float16*)alloc(DH_H * 2);
    float*    pre_e = (float*)alloc(PRE_F * 4);
    float*    pre_d = (float*)alloc(PRE_F * 4);

    // 1) Gather + pack activations
    pack_x_kernel<<<(int)(XE_H / 256), 256, 0, stream>>>(enc_ids, E, xe, TX, (int)XE_H);
    pack_x_kernel<<<(int)(XE_H / 256), 256, 0, stream>>>(dec_ids, E, xd, TY, (int)XE_H);

    // 2) Pack weights (Wx = rows [0,256), Wh = rows [256,768) of [768,2048])
    pack_b_kernel<<<(int)(WX_H / 256), 256, 0, stream>>>(W_enc,                    wxe, NT_G, G4H, (int)WX_H);
    pack_b_kernel<<<(int)(WH_H / 256), 256, 0, stream>>>(W_enc + (size_t)Dd * G4H, whe, NT_G, G4H, (int)WH_H);
    pack_b_kernel<<<(int)(WX_H / 256), 256, 0, stream>>>(W_dec,                    wxd, NT_G, G4H, (int)WX_H);
    pack_b_kernel<<<(int)(WH_H / 256), 256, 0, stream>>>(W_dec + (size_t)Dd * G4H, whd, NT_G, G4H, (int)WH_H);
    pack_b_kernel<<<(int)(WO_H / 256), 256, 0, stream>>>(W_out,                    wout, NT_V, Vv, (int)WO_H);

    // 3) pre[t] = x_t @ Wx + b  (parallel over all timesteps)
    gemm_pre_kernel<<<1024, 256, 0, stream>>>(xe, wxe, b_enc, pre_e, 256);
    gemm_pre_kernel<<<1024, 256, 0, stream>>>(xd, wxd, b_dec, pre_d, 256);

    // 4) Sequential recurrence (encoder then decoder) on one WGP
    lstm_seq_kernel<<<1, 1024, 0, stream>>>(pre_e, pre_d, whe, whd, dech, TX, TY);

    // 5) Output projection
    proj_kernel<<<16000, 256, 0, stream>>>(dech, wout, b_out, out);
}
